// QuantMLP_23948737643282
// MI455X (gfx1250) — 8-bit matrix operand order repaired and checked by reading, compiled, not run on hardware
//
#include <hip/hip_runtime.h>
#include <hip/hip_bf16.h>
#include <stdint.h>


typedef __attribute__((ext_vector_type(8))) int v8i;

#define MLP_D 768
#define MLP_H 3072

#define WS_ROWSUM1   64
#define WS_ROWSUM2   (WS_ROWSUM1 + 12288)
#define WS_QX        16384

__device__ inline unsigned enc_f32(float f) {
  unsigned u = __float_as_uint(f);
  return (u & 0x80000000u) ? ~u : (u | 0x80000000u);
}
__device__ inline float dec_f32(unsigned u) {
  unsigned b = (u & 0x80000000u) ? (u & 0x7FFFFFFFu) : ~u;
  return __uint_as_float(b);
}
__device__ inline unsigned umin32(unsigned a, unsigned b) { return a < b ? a : b; }
__device__ inline unsigned umax32(unsigned a, unsigned b) { return a > b ? a : b; }

__device__ inline unsigned short f32_to_bf16(float f) {
  unsigned u = __float_as_uint(f);
  unsigned r = 0x7FFFu + ((u >> 16) & 1u);
  return (unsigned short)((u + r) >> 16);
}

__device__ inline float int_gelu(float x) {
  const float t  = x * 0.70710678118654752f;
  const float at = fminf(fabsf(t), 1.769f);
  const float d  = at - 1.769f;
  const float L  = copysignf(-0.2888f * d * d + 1.0f, t);
  return x * 0.5f * (1.0f + L);
}

__device__ __forceinline__ void async_b128(unsigned lds_byte_off,
                                           unsigned gl_byte_off,
                                           const void* gl_base) {
  asm volatile("global_load_async_to_lds_b128 %0, %1, %2"
               :: "v"(lds_byte_off), "v"(gl_byte_off), "s"(gl_base)
               : "memory");
}
__device__ __forceinline__ void wait_async_zero() {
#if __has_builtin(__builtin_amdgcn_s_wait_asynccnt)
  __builtin_amdgcn_s_wait_asynccnt(0);
#else
  asm volatile("s_wait_asynccnt 0x0" ::: "memory");
#endif
}

__global__ void k_init_scalars(unsigned* s) {
  if (threadIdx.x == 0) {
    s[0] = 0xFFFFFFFFu; s[1] = 0u;
    s[2] = 0u;          s[3] = 0u;
    s[4] = 0xFFFFFFFFu; s[5] = 0u;
  }
}

__global__ __launch_bounds__(256) void k_minmax_f32(const float* __restrict__ x,
                                                    int n4, unsigned* __restrict__ mm) {
  unsigned emin = 0xFFFFFFFFu, emax = 0u;
  const float4* xv = reinterpret_cast<const float4*>(x);
  for (int i = blockIdx.x * blockDim.x + threadIdx.x; i < n4; i += gridDim.x * blockDim.x) {
    float4 v = xv[i];
    emin = umin32(emin, umin32(umin32(enc_f32(v.x), enc_f32(v.y)),
                               umin32(enc_f32(v.z), enc_f32(v.w))));
    emax = umax32(emax, umax32(umax32(enc_f32(v.x), enc_f32(v.y)),
                               umax32(enc_f32(v.z), enc_f32(v.w))));
  }
  for (int off = 16; off; off >>= 1) {
    emin = umin32(emin, (unsigned)__shfl_xor((int)emin, off, 32));
    emax = umax32(emax, (unsigned)__shfl_xor((int)emax, off, 32));
  }
  __shared__ unsigned smn[8], smx[8];
  int wave = threadIdx.x >> 5, lane = threadIdx.x & 31;
  if (lane == 0) { smn[wave] = emin; smx[wave] = emax; }
  __syncthreads();
  if (threadIdx.x == 0) {
    unsigned mn = smn[0], mx = smx[0];
    for (int i = 1; i < 8; ++i) { mn = umin32(mn, smn[i]); mx = umax32(mx, smx[i]); }
    atomicMin(&mm[0], mn);
    atomicMax(&mm[1], mx);
  }
}

__global__ __launch_bounds__(256) void k_absmax_f32(const float* __restrict__ w,
                                                    int n4, unsigned* __restrict__ out) {
  float a = 0.0f;
  const float4* wv = reinterpret_cast<const float4*>(w);
  for (int i = blockIdx.x * blockDim.x + threadIdx.x; i < n4; i += gridDim.x * blockDim.x) {
    float4 v = wv[i];
    a = fmaxf(a, fmaxf(fmaxf(fabsf(v.x), fabsf(v.y)), fmaxf(fabsf(v.z), fabsf(v.w))));
  }
  for (int off = 16; off; off >>= 1)
    a = fmaxf(a, __shfl_xor(a, off, 32));
  __shared__ float sa[8];
  int wave = threadIdx.x >> 5, lane = threadIdx.x & 31;
  if (lane == 0) sa[wave] = a;
  __syncthreads();
  if (threadIdx.x == 0) {
    float m = sa[0];
    for (int i = 1; i < 8; ++i) m = fmaxf(m, sa[i]);
    atomicMax(out, __float_as_uint(m));
  }
}

__global__ __launch_bounds__(256) void k_quant_weight(const float* __restrict__ w,
                                                      signed char* __restrict__ wq,
                                                      int* __restrict__ rowsum,
                                                      const unsigned* __restrict__ absmax_bits,
                                                      int K) {
  const int row = blockIdx.x;
  const float s = fmaxf(__uint_as_float(*absmax_bits), 1e-8f) * (1.0f / 127.0f);
  const float* wr = w + (size_t)row * K;
  signed char* qr = wq + (size_t)row * K;
  int sum = 0;
  for (int c = threadIdx.x; c < K; c += blockDim.x) {
    float q = fminf(fmaxf(rintf(wr[c] / s), -128.0f), 127.0f);
    int qi = (int)q;
    qr[c] = (signed char)qi;
    sum += qi;
  }
  for (int off = 16; off; off >>= 1) sum += __shfl_xor(sum, off, 32);
  __shared__ int ss[8];
  int wave = threadIdx.x >> 5, lane = threadIdx.x & 31;
  if (lane == 0) ss[wave] = sum;
  __syncthreads();
  if (threadIdx.x == 0) {
    int t = 0;
    for (int i = 0; i < 8; ++i) t += ss[i];
    rowsum[row] = t;
  }
}

__global__ __launch_bounds__(256) void k_quant_x(const float* __restrict__ x,
                                                 unsigned char* __restrict__ qx,
                                                 const unsigned* __restrict__ scal, int n4) {
  const float xmin = dec_f32(scal[0]);
  const float xmax = dec_f32(scal[1]);
  const float s  = fmaxf(xmax - xmin, 1e-8f) * (1.0f / 255.0f);
  const float zp = rintf(-xmin / s);
  const float4* xv = reinterpret_cast<const float4*>(x);
  unsigned* qv = reinterpret_cast<unsigned*>(qx);
  for (int i = blockIdx.x * blockDim.x + threadIdx.x; i < n4; i += gridDim.x * blockDim.x) {
    float4 v = xv[i];
    unsigned q0 = (unsigned)fminf(fmaxf(rintf(v.x / s) + zp, 0.0f), 255.0f);
    unsigned q1 = (unsigned)fminf(fmaxf(rintf(v.y / s) + zp, 0.0f), 255.0f);
    unsigned q2 = (unsigned)fminf(fmaxf(rintf(v.z / s) + zp, 0.0f), 255.0f);
    unsigned q3 = (unsigned)fminf(fmaxf(rintf(v.w / s) + zp, 0.0f), 255.0f);
    qv[i] = q0 | (q1 << 8) | (q2 << 16) | (q3 << 24);
  }
}

__global__ __launch_bounds__(256) void k_quant_g(const unsigned short* __restrict__ g,
                                                 unsigned char* __restrict__ qg,
                                                 const unsigned* __restrict__ scal, int n8) {
  const float gmin = dec_f32(scal[4]);
  const float gmax = dec_f32(scal[5]);
  const float s  = fmaxf(gmax - gmin, 1e-8f) * (1.0f / 255.0f);
  const float zp = rintf(-gmin / s);
  const uint4* gv = reinterpret_cast<const uint4*>(g);
  uint2* qv = reinterpret_cast<uint2*>(qg);
  for (int i = blockIdx.x * blockDim.x + threadIdx.x; i < n8; i += gridDim.x * blockDim.x) {
    uint4 u = gv[i];
    unsigned w[4] = {u.x, u.y, u.z, u.w};
    unsigned lo = 0, hi = 0;
    #pragma unroll
    for (int j = 0; j < 4; ++j) {
      float f0 = __uint_as_float(w[j] << 16);
      float f1 = __uint_as_float(w[j] & 0xFFFF0000u);
      unsigned q0 = (unsigned)fminf(fmaxf(rintf(f0 / s) + zp, 0.0f), 255.0f);
      unsigned q1 = (unsigned)fminf(fmaxf(rintf(f1 / s) + zp, 0.0f), 255.0f);
      unsigned pk = q0 | (q1 << 8);
      if (j < 2) lo |= pk << (16 * j);
      else       hi |= pk << (16 * (j - 2));
    }
    uint2 r; r.x = lo; r.y = hi;
    qv[i] = r;
  }
}

#define SLAB_BYTES 20480
#define SLAB_BOFF  4096

__device__ __forceinline__ v8i lds_a_frag(const char* slab, int rowtile,
                                          int lanelo, int laneHi) {
  const char* p = slab + (rowtile * 16 + lanelo) * 64 + (laneHi << 4);
  union { v8i v; unsigned long long u[4]; } a;
  #pragma unroll
  for (int c = 0; c < 4; ++c)
    a.u[c] = *reinterpret_cast<const unsigned long long*>(p + (c >> 1) * 32 + (c & 1) * 8);
  return a.v;
}
__device__ __forceinline__ v8i lds_b_frag(const char* slab, int col,
                                          int laneHi) {
  const char* p = slab + SLAB_BOFF + col * 64 + (laneHi << 4);
  union { v8i v; uint4 q[2]; } b;
  b.q[0] = *reinterpret_cast<const uint4*>(p);
  b.q[1] = *reinterpret_cast<const uint4*>(p + 32);
  return b.v;
}

__global__ __launch_bounds__(256) void k_gemm1_gelu(
    const unsigned char* __restrict__ qx,
    const signed char*  __restrict__ w1q,
    const int*          __restrict__ rowsum1,
    const float*        __restrict__ b1,
    const unsigned*     __restrict__ scal,
    unsigned short*     __restrict__ g)
{
  const int K = MLP_D, N = MLP_H;
  const int tid  = threadIdx.x;
  const int lane = tid & 31;
  const int wave = tid >> 5;
  const int laneHi = lane >> 4;
  const int lanelo = lane & 15;
  const int rowtile = wave & 3;
  const int colgrp  = wave >> 2;

  const int m0blk = blockIdx.x * 64;
  const int n0blk = blockIdx.y * 256;
  const int m0 = m0blk + rowtile * 16;
  const int n0 = n0blk + colgrp * 128;

  const float xmin = dec_f32(scal[0]);
  const float xmax = dec_f32(scal[1]);
  const float sx  = fmaxf(xmax - xmin, 1e-8f) * (1.0f / 255.0f);
  const float zpx = rintf(-xmin / sx);
  const float sw  = fmaxf(__uint_as_float(scal[2]), 1e-8f) * (1.0f / 127.0f);
  const float scale = sx * sw;

  __shared__ __align__(16) char smem[2 * SLAB_BYTES];
  const unsigned lds_base = (unsigned)(uintptr_t)&smem[0];

  const unsigned a_lds  = (unsigned)((tid >> 2) * 64 + (tid & 3) * 16);
  const unsigned a_goff = (unsigned)((m0blk + (tid >> 2)) * K + (tid & 3) * 16);
  const unsigned b_lds  = (unsigned)(SLAB_BOFF + tid * 64);
  const unsigned b_goff = (unsigned)((n0blk + tid) * K);

  v8i acc[8];
  #pragma unroll
  for (int t = 0; t < 8; ++t) acc[t] = (v8i)(0);

  const int nk = K / 64;
  {
    async_b128(lds_base + a_lds, a_goff, qx);
    #pragma unroll
    for (int c = 0; c < 4; ++c)
      async_b128(lds_base + b_lds + c * 16, b_goff + c * 16, w1q);
  }

  for (int i = 0; i < nk; ++i) {
    wait_async_zero();
    __syncthreads();
    if (i + 1 < nk) {
      const unsigned nb = lds_base + ((i + 1) & 1) * SLAB_BYTES;
      const unsigned k1 = (unsigned)((i + 1) * 64);
      async_b128(nb + a_lds, a_goff + k1, qx);
      #pragma unroll
      for (int c = 0; c < 4; ++c)
        async_b128(nb + b_lds + c * 16, b_goff + k1 + c * 16, w1q);
    }
    const char* slab = smem + (i & 1) * SLAB_BYTES;
    v8i a = lds_a_frag(slab, rowtile, lanelo, laneHi);
    #pragma unroll
    for (int t = 0; t < 8; ++t) {
      v8i b = lds_b_frag(slab, colgrp * 128 + t * 16 + lanelo, laneHi);
      acc[t] = __builtin_amdgcn_wmma_i32_16x16x64_iu8(false, a, true, b, acc[t],
                                                      false, false);
    }
  }

  unsigned emin = 0xFFFFFFFFu, emax = 0u;
  #pragma unroll
  for (int t = 0; t < 8; ++t) {
    const int n = n0 + t * 16 + lanelo;
    const float corr = zpx * (float)rowsum1[n];
    const float bias = b1[n];
    #pragma unroll
    for (int r = 0; r < 8; ++r) {
      const int m = m0 + r + (laneHi << 3);
      float h  = scale * ((float)acc[t][r] - corr) + bias;
      float gv = int_gelu(h);
      unsigned e = enc_f32(gv);
      emin = umin32(emin, e);
      emax = umax32(emax, e);
      g[(size_t)m * N + n] = f32_to_bf16(gv);
    }
  }

  for (int off = 16; off; off >>= 1) {
    emin = umin32(emin, (unsigned)__shfl_xor((int)emin, off, 32));
    emax = umax32(emax, (unsigned)__shfl_xor((int)emax, off, 32));
  }
  __shared__ unsigned smn[8], smx[8];
  if (lane == 0) { smn[wave] = emin; smx[wave] = emax; }
  __syncthreads();
  if (tid == 0) {
    unsigned mn = smn[0], mx = smx[0];
    for (int i = 1; i < 8; ++i) { mn = umin32(mn, smn[i]); mx = umax32(mx, smx[i]); }
    atomicMin(const_cast<unsigned*>(&scal[4]), mn);
    atomicMax(const_cast<unsigned*>(&scal[5]), mx);
  }
}

__global__ __launch_bounds__(256) void k_gemm2(
    const unsigned char* __restrict__ qg,
    const signed char*  __restrict__ w2q,
    const int*          __restrict__ rowsum2,
    const float*        __restrict__ b2,
    const unsigned*     __restrict__ scal,
    float*              __restrict__ out)
{
  const int K = MLP_H, N = MLP_D;
  const int tid  = threadIdx.x;
  const int lane = tid & 31;
  const int wave = tid >> 5;
  const int laneHi = lane >> 4;
  const int lanelo = lane & 15;
  const int rowtile = wave & 3;
  const int colgrp  = wave >> 2;

  const int m0blk = blockIdx.x * 64;
  const int n0blk = blockIdx.y * 256;
  const int m0 = m0blk + rowtile * 16;
  const int n0 = n0blk + colgrp * 128;

  const float gmin = dec_f32(scal[4]);
  const float gmax = dec_f32(scal[5]);
  const float sg  = fmaxf(gmax - gmin, 1e-8f) * (1.0f / 255.0f);
  const float zpg = rintf(-gmin / sg);
  const float sw  = fmaxf(__uint_as_float(scal[3]), 1e-8f) * (1.0f / 127.0f);
  const float scale = sg * sw;

  __shared__ __align__(16) char smem[2 * SLAB_BYTES];
  const unsigned lds_base = (unsigned)(uintptr_t)&smem[0];

  const unsigned a_lds  = (unsigned)((tid >> 2) * 64 + (tid & 3) * 16);
  const unsigned a_goff = (unsigned)((m0blk + (tid >> 2)) * K + (tid & 3) * 16);
  const unsigned b_lds  = (unsigned)(SLAB_BOFF + tid * 64);
  const unsigned b_goff = (unsigned)((n0blk + tid) * K);

  v8i acc[8];
  #pragma unroll
  for (int t = 0; t < 8; ++t) acc[t] = (v8i)(0);

  const int nk = K / 64;
  {
    async_b128(lds_base + a_lds, a_goff, qg);
    #pragma unroll
    for (int c = 0; c < 4; ++c)
      async_b128(lds_base + b_lds + c * 16, b_goff + c * 16, w2q);
  }

  for (int i = 0; i < nk; ++i) {
    wait_async_zero();
    __syncthreads();
    if (i + 1 < nk) {
      const unsigned nb = lds_base + ((i + 1) & 1) * SLAB_BYTES;
      const unsigned k1 = (unsigned)((i + 1) * 64);
      async_b128(nb + a_lds, a_goff + k1, qg);
      #pragma unroll
      for (int c = 0; c < 4; ++c)
        async_b128(nb + b_lds + c * 16, b_goff + k1 + c * 16, w2q);
    }
    const char* slab = smem + (i & 1) * SLAB_BYTES;
    v8i a = lds_a_frag(slab, rowtile, lanelo, laneHi);
    #pragma unroll
    for (int t = 0; t < 8; ++t) {
      v8i b = lds_b_frag(slab, colgrp * 128 + t * 16 + lanelo, laneHi);
      acc[t] = __builtin_amdgcn_wmma_i32_16x16x64_iu8(false, a, true, b, acc[t],
                                                      false, false);
    }
  }

  #pragma unroll
  for (int t = 0; t < 8; ++t) {
    const int n = n0 + t * 16 + lanelo;
    const float corr = zpg * (float)rowsum2[n];
    const float bias = b2[n];
    #pragma unroll
    for (int r = 0; r < 8; ++r) {
      const int m = m0 + r + (laneHi << 3);
      out[(size_t)m * N + n] = scale * ((float)acc[t][r] - corr) + bias;
    }
  }
}

extern "C" void kernel_launch(void* const* d_in, const int* in_sizes, int n_in,
                              void* d_out, int out_size, void* d_ws, size_t ws_size,
                              hipStream_t stream) {
  const float* x  = (const float*)d_in[0];
  const float* w1 = (const float*)d_in[1];
  const float* b1 = (const float*)d_in[2];
  const float* w2 = (const float*)d_in[3];
  const float* b2 = (const float*)d_in[4];
  float* out = (float*)d_out;

  const int M = in_sizes[0] / MLP_D;

  char* ws = (char*)d_ws;
  unsigned* scal    = (unsigned*)ws;
  int*      rowsum1 = (int*)(ws + WS_ROWSUM1);
  int*      rowsum2 = (int*)(ws + WS_ROWSUM2);
  unsigned char* qx = (unsigned char*)(ws + WS_QX);
  size_t off = WS_QX + (size_t)M * MLP_D;
  signed char* w1q = (signed char*)(ws + off);  off += (size_t)MLP_H * MLP_D;
  signed char* w2q = (signed char*)(ws + off);  off += (size_t)MLP_D * MLP_H;
  unsigned short* g = (unsigned short*)(ws + off); off += (size_t)M * MLP_H * 2;
  unsigned char* qg = (unsigned char*)(ws + off);

  const int nx = M * MLP_D;
  const int ng = M * MLP_H;

  k_init_scalars<<<1, 64, 0, stream>>>(scal);
  k_minmax_f32<<<1024, 256, 0, stream>>>(x, nx / 4, scal);
  k_absmax_f32<<<512, 256, 0, stream>>>(w1, (MLP_H * MLP_D) / 4, &scal[2]);
  k_absmax_f32<<<512, 256, 0, stream>>>(w2, (MLP_D * MLP_H) / 4, &scal[3]);
  k_quant_weight<<<MLP_H, 256, 0, stream>>>(w1, w1q, rowsum1, &scal[2], MLP_D);
  k_quant_weight<<<MLP_D, 256, 0, stream>>>(w2, w2q, rowsum2, &scal[3], MLP_H);
  k_quant_x<<<2048, 256, 0, stream>>>(x, qx, scal, nx / 4);

  dim3 g1(M / 64, MLP_H / 256);
  k_gemm1_gelu<<<g1, 256, 0, stream>>>(qx, w1q, rowsum1, b1, scal, g);

  k_quant_g<<<4096, 256, 0, stream>>>(g, qg, scal, ng / 8);

  dim3 g2(M / 64, MLP_D / 256);
  k_gemm2<<<g2, 256, 0, stream>>>(qg, w2q, rowsum2, b2, scal, out);
}
